// RMultiHeadSelfAttention_87995289960719
// MI455X (gfx1250) — matrix operand order repaired and checked by reading, not compiled, not run on hardware
//
#include <hip/hip_runtime.h>
#include <hip/hip_bf16.h>

// MI455X / gfx1250, wave32. MHA: B=2, S=2048, D=1024, H=16, dk=64.
// ~69 GFLOP vs ~50MB traffic (resident in 192MB L2) -> compute bound ->
// bf16 WMMA for every GEMM, fp32 softmax/RoPE on accumulators.
// Round 3: GEMM K-loop ping-pong double buffering (2x unroll) so the
// software pipeline needs no register rotation (kills the v_dual_mov storm);
// attention keeps TDM-staged, double-buffered K/V tiles in LDS.

typedef __attribute__((ext_vector_type(16))) __bf16 v16bf;
typedef __attribute__((ext_vector_type(8)))  __bf16 v8bf;
typedef __attribute__((ext_vector_type(4)))  __bf16 v4bf;
typedef __attribute__((ext_vector_type(8)))  float  v8f;

#define DMODEL 1024
#define SLEN   2048
#define NB     2
#define NH     16
#define DK     64

#ifndef __has_builtin
#define __has_builtin(x) 0
#endif
#if __has_builtin(__builtin_amdgcn_tensor_load_to_lds) && \
    __has_builtin(__builtin_amdgcn_s_wait_tensorcnt)
#define USE_TDM 1
#else
#define USE_TDM 0
#endif

union ABu { v16bf v; v8bf h[2]; };

__device__ __forceinline__ v8f wmma_bf16(v16bf a, v16bf b, v8f c) {
  return __builtin_amdgcn_wmma_f32_16x16x32_bf16(false, a, false, b,
                                                 (short)0, c, false, false);
}

// ---- WMMA operand loaders --------------------------------------------------
// A operand 16x32 (MxK): lane l: m=l&15, hi=l>>4; elem i: k=hi*16+i
__device__ __forceinline__ v16bf load_a_16x32(const __bf16* __restrict__ base,
                                              int ld, int m0, int k0) {
  const int lane = threadIdx.x & 31;
  const __bf16* row = base + (size_t)(m0 + (lane & 15)) * ld
                           + (k0 + ((lane >> 4) << 4));
  ABu u;
  u.h[0] = *(const v8bf*)(row);
  u.h[1] = *(const v8bf*)(row + 8);
  return u.v;
}

// B operand 32x16 (KxN), B[k][n] = base[n*ld + k]: lane l: n=l&15, hi=l>>4,
// elem i: k = hi*16 + i  -> one contiguous 32B chunk per lane.
__device__ __forceinline__ v16bf load_b_kmajor(const __bf16* __restrict__ base,
                                               int ld, int n0, int k0) {
  const int lane = threadIdx.x & 31;
  const __bf16* p = base + (size_t)(n0 + (lane & 15)) * ld
                         + (k0 + ((lane >> 4) << 4));
  return *(const v16bf*)p;
}

// Same B layout but from LDS (addrspace inferred -> ds_load_b128 pairs).
__device__ __forceinline__ v16bf load_b_lds(const __bf16* base, int ld,
                                            int n0, int k0) {
  const int lane = threadIdx.x & 31;
  return *(const v16bf*)(base + (n0 + (lane & 15)) * ld
                              + (k0 + ((lane >> 4) << 4)));
}

// ---- Tensor Data Mover: 2D tile Global->LDS (D# per ISA 8.3/8.4) -----------
#if USE_TDM
typedef __attribute__((ext_vector_type(4))) unsigned int u32x4;
typedef __attribute__((ext_vector_type(4))) int i32x4;
typedef __attribute__((ext_vector_type(8))) int i32x8;

__device__ __forceinline__ unsigned lds_u32(const void* p) {
  // generic LDS address: low 32 bits are the LDS byte offset (aperture scheme)
  return (unsigned)(unsigned long long)p;
}

// data_size = 2 bytes (bf16). tile_w contiguous elems, tile_h rows of
// row_stride elems; tensor_w/h for OOB checks.
__device__ __forceinline__ void tdm_load_2d(const void* gptr, unsigned lds_addr,
                                            unsigned tile_w, unsigned tile_h,
                                            unsigned tensor_w, unsigned tensor_h,
                                            unsigned row_stride) {
  unsigned long long ga = (unsigned long long)gptr;
  u32x4 g0;
  g0[0] = 1u;                                                // count=1, user D#
  g0[1] = lds_addr;                                          // LDS byte addr
  g0[2] = (unsigned)ga;                                      // global[31:0]
  g0[3] = (unsigned)((ga >> 32) & 0x01FFFFFFu) | (2u << 30); // [56:32]|type=2
  i32x8 g1;
  g1[0] = (int)(1u << 16);                                   // data_size=1 (2B)
  g1[1] = (int)((tensor_w & 0xFFFFu) << 16);                 // tdim0[15:0]
  g1[2] = (int)(((tensor_w >> 16) & 0xFFFFu) | ((tensor_h & 0xFFFFu) << 16));
  g1[3] = (int)(((tensor_h >> 16) & 0xFFFFu) | ((tile_w & 0xFFFFu) << 16));
  g1[4] = (int)(tile_h & 0xFFFFu);                           // tile_dim1, dim2=0
  g1[5] = (int)row_stride;                                   // dim0_stride[31:0]
  g1[6] = 0;
  g1[7] = 0;
  i32x4 gz = {0, 0, 0, 0};
#if __clang_major__ >= 23
  i32x8 gz8 = {0, 0, 0, 0, 0, 0, 0, 0};
  __builtin_amdgcn_tensor_load_to_lds(g0, g1, gz, gz, gz8, 0);
#else
  __builtin_amdgcn_tensor_load_to_lds(g0, g1, gz, gz, 0);
#endif
}
#endif  // USE_TDM

// ---- fp32 -> bf16 conversion ----------------------------------------------
__global__ void __launch_bounds__(256)
cvt_f32_bf16_x4(const float4* __restrict__ in, v4bf* __restrict__ out, int n4) {
  int i = blockIdx.x * blockDim.x + threadIdx.x;
  if (i < n4) {
    float4 f = in[i];
    v4bf o;
    o[0] = (__bf16)f.x; o[1] = (__bf16)f.y;
    o[2] = (__bf16)f.z; o[3] = (__bf16)f.w;
    out[i] = o;
  }
}

// ---- GEMM: Y[m][n] = sum_k X[m][k] * W[n][k]  (x @ w^T) --------------------
// Wave = 16(M) x 64(N) strip, K step 32, 2x-unrolled ping-pong pipeline:
// even steps use {a0,b0}, odd steps {a1,b1}; each load writes directly into
// the registers its WMMA consumes (no rotation movs, partial loadcnt waits).
// mode 0/1: Q/K proj + RoPE -> bf16 [B,H,S,DK];  mode 2: V -> bf16 [B,H,DK,S]
// mode 3: output proj -> fp32 [M,DMODEL]
__global__ void __launch_bounds__(256)
gemm_wmma(const __bf16* __restrict__ X, const __bf16* __restrict__ W,
          void* __restrict__ outp, const int* __restrict__ pos, int mode) {
  const int wave = blockIdx.x * (blockDim.x >> 5) + (threadIdx.x >> 5);
  const int lane = threadIdx.x & 31;
  const int nblk = wave & 15;
  const int mblk = wave >> 4;
  const int m0 = mblk << 4;
  const int n0 = nblk << 6;

  v8f acc[4] = {};
  v16bf a0 = load_a_16x32(X, DMODEL, m0, 0);
  v16bf b0[4];
#pragma unroll
  for (int t = 0; t < 4; ++t) b0[t] = load_b_kmajor(W, DMODEL, n0 + t * 16, 0);

  for (int s = 0; s < (DMODEL / 64) - 1; ++s) {   // 15 iterations
    const int k1 = s * 64 + 32;
    const int k2 = s * 64 + 64;
    v16bf a1 = load_a_16x32(X, DMODEL, m0, k1);
    v16bf b1[4];
#pragma unroll
    for (int t = 0; t < 4; ++t) b1[t] = load_b_kmajor(W, DMODEL, n0 + t * 16, k1);
#pragma unroll
    for (int t = 0; t < 4; ++t) acc[t] = wmma_bf16(a0, b0[t], acc[t]);
    a0 = load_a_16x32(X, DMODEL, m0, k2);
#pragma unroll
    for (int t = 0; t < 4; ++t) b0[t] = load_b_kmajor(W, DMODEL, n0 + t * 16, k2);
#pragma unroll
    for (int t = 0; t < 4; ++t) acc[t] = wmma_bf16(a1, b1[t], acc[t]);
  }
  {  // epilogue: steps 30 (in a0/b0) and 31
    v16bf a1 = load_a_16x32(X, DMODEL, m0, DMODEL - 32);
    v16bf b1[4];
#pragma unroll
    for (int t = 0; t < 4; ++t)
      b1[t] = load_b_kmajor(W, DMODEL, n0 + t * 16, DMODEL - 32);
#pragma unroll
    for (int t = 0; t < 4; ++t) acc[t] = wmma_bf16(a0, b0[t], acc[t]);
#pragma unroll
    for (int t = 0; t < 4; ++t) acc[t] = wmma_bf16(a1, b1[t], acc[t]);
  }

  const int hi = lane >> 4;
  const int nl = lane & 15;

  if (mode == 3) {
    float* out = (float*)outp;
#pragma unroll
    for (int t = 0; t < 4; ++t) {
      int n = n0 + t * 16 + nl;
#pragma unroll
      for (int r = 0; r < 8; ++r) {
        int m = m0 + r + hi * 8;
        out[(size_t)m * DMODEL + n] = acc[t][r];
      }
    }
    return;
  }

  __bf16* out = (__bf16*)outp;
  const bool do_rope = (mode < 2);
#pragma unroll
  for (int t = 0; t < 4; ++t) {
    int n = n0 + t * 16 + nl;
    int h = n >> 6;
    int j = n & (DK - 1);
    bool even = ((j & 1) == 0);
    // inv_freq[i] = theta^(-2i/dk) = exp(-i * ln(10000)/32)
    float invf = do_rope ? __expf(-(float)(j >> 1) * 0.2878231366f) : 0.0f;
#pragma unroll
    for (int r = 0; r < 8; ++r) {
      int m = m0 + r + hi * 8;
      int bb = m >> 11;
      int s = m & (SLEN - 1);
      float v = acc[t][r];
      if (do_rope) {
        float partner = __shfl_xor(v, 1, 32);   // paired head-dim in lane^1
        float ang = (float)pos[s] * invf;
        float sn, cs;
        __sincosf(ang, &sn, &cs);
        v = even ? (v * cs - partner * sn) : (partner * sn + v * cs);
      }
      if (mode == 2) {   // V^T: [b][h][j][s]
        out[(((size_t)bb * NH + h) * DK + j) * SLEN + s] = (__bf16)v;
      } else {           // Q/K: [b][h][s][j]
        out[(((size_t)bb * NH + h) * SLEN + s) * DK + j] = (__bf16)v;
      }
    }
  }
}

// ---- Flash attention (causal), TDM-staged K/V ------------------------------
// Block = 128 thr = 4 waves, one (b,h) per block, wave w owns q-tile q0=..+w*16.
// Per 32-key iteration: wave0 TDM-loads K tile [32xDK] and V^T tile [DKx32]
// into double-buffered LDS (overlapped with compute), QK^T = 4 WMMA from LDS,
// online softmax fp32, P via LDS relayout, PV = 4 WMMA from LDS.
__global__ void __launch_bounds__(128)
attn_flash(const __bf16* __restrict__ Q, const __bf16* __restrict__ K,
           const __bf16* __restrict__ Vt, __bf16* __restrict__ AO) {
  __shared__ __align__(32) __bf16 Kt_s[2][32 * DK];  // [buf][key][d]
  __shared__ __align__(32) __bf16 Vs[2][DK * 32];    // [buf][d][key]
  __shared__ __align__(16) __bf16 Ps[4][16][32];     // per-wave P tile

  const int w    = threadIdx.x >> 5;
  const int lane = threadIdx.x & 31;
  const int hi   = lane >> 4;
  const int nl   = lane & 15;
  const int qblk = blockIdx.x & 31;   // S/64
  const int bh   = blockIdx.x >> 5;   // B*H
  const int q0   = qblk * 64 + w * 16;

  const __bf16* Qp = Q  + (size_t)bh * SLEN * DK;
  const __bf16* Kp = K  + (size_t)bh * SLEN * DK;
  const __bf16* Vp = Vt + (size_t)bh * DK * SLEN;

  v16bf aq0 = load_a_16x32(Qp, DK, q0, 0);
  v16bf aq1 = load_a_16x32(Qp, DK, q0, 32);

  v8f o[4] = {};
  float mrow[8], lrow[8];
#pragma unroll
  for (int r = 0; r < 8; ++r) { mrow[r] = -1e30f; lrow[r] = 0.0f; }

  const int nit = qblk * 2 + 2;  // block-uniform causal extent (32 keys/iter)

  auto compute_tile = [&](int cur, int key0) {
    float pl[2][8];
#pragma unroll
    for (int t = 0; t < 2; ++t) {
      v8f sc = {};
      v16bf bk0 = load_b_lds(Kt_s[cur], DK, t * 16, 0);
      sc = wmma_bf16(aq0, bk0, sc);
      v16bf bk1 = load_b_lds(Kt_s[cur], DK, t * 16, 32);
      sc = wmma_bf16(aq1, bk1, sc);
      int kc = key0 + t * 16 + nl;
#pragma unroll
      for (int r = 0; r < 8; ++r) {
        int qr = q0 + r + hi * 8;
        float sv = sc[r] * 0.125f;             // 1/sqrt(dk)
        pl[t][r] = (kc <= qr) ? sv : -1e30f;   // causal (finite: no NaN)
      }
    }
#pragma unroll
    for (int r = 0; r < 8; ++r) {
      float mx = fmaxf(pl[0][r], pl[1][r]);
#pragma unroll
      for (int off = 1; off < 16; off <<= 1)
        mx = fmaxf(mx, __shfl_xor(mx, off, 32));
      float mnew  = fmaxf(mrow[r], mx);
      float alpha = __expf(mrow[r] - mnew);
      float p0 = __expf(pl[0][r] - mnew);
      float p1 = __expf(pl[1][r] - mnew);
      Ps[w][r + hi * 8][nl]      = (__bf16)p0;
      Ps[w][r + hi * 8][16 + nl] = (__bf16)p1;
      float rs = p0 + p1;
#pragma unroll
      for (int off = 1; off < 16; off <<= 1)
        rs += __shfl_xor(rs, off, 32);
      lrow[r] = lrow[r] * alpha + rs;
      mrow[r] = mnew;
#pragma unroll
      for (int nt = 0; nt < 4; ++nt) o[nt][r] *= alpha;
    }
    // P (16x32) back in A-operand layout: 2x ds_load_b128 per lane
    const __bf16* prow = &Ps[w][nl][hi * 16];
    ABu u;
    u.h[0] = *(const v8bf*)(prow);
    u.h[1] = *(const v8bf*)(prow + 8);
    v16bf ap = u.v;
#pragma unroll
    for (int nt = 0; nt < 4; ++nt) {
      v16bf bv = load_b_lds(Vs[cur], 32, nt * 16, 0);
      o[nt] = wmma_bf16(ap, bv, o[nt]);
    }
  };

#if USE_TDM
  auto issue_tiles = [&](int it) {
    int key0 = it * 32;
    int cur = it & 1;
    // K tile: 32 rows of DK contiguous bf16
    tdm_load_2d(Kp + (size_t)key0 * DK, lds_u32(Kt_s[cur]),
                /*tile_w=*/DK, /*tile_h=*/32,
                /*tensor_w=*/DK, /*tensor_h=*/SLEN, /*row_stride=*/DK);
    // V^T tile: DK rows of 32 contiguous keys, row stride SLEN
    tdm_load_2d(Vp + key0, lds_u32(Vs[cur]),
                /*tile_w=*/32, /*tile_h=*/DK,
                /*tensor_w=*/SLEN, /*tensor_h=*/DK, /*row_stride=*/SLEN);
  };
  if (w == 0) issue_tiles(0);
  for (int it = 0; it < nit; ++it) {
    int cur = it & 1;
    int key0 = it * 32;
    if (w == 0) __builtin_amdgcn_s_wait_tensorcnt(0);
    __syncthreads();                       // tiles visible to all waves
    if (w == 0 && it + 1 < nit) issue_tiles(it + 1);  // overlap next DMA
    compute_tile(cur, key0);
  }
#else
  for (int it = 0; it < nit; ++it) {
    int cur = it & 1;
    int key0 = it * 32;
    __syncthreads();
    {  // cooperative copy fallback (no TDM builtin on this toolchain)
      const unsigned* gk = (const unsigned*)(Kp + (size_t)key0 * DK);
      unsigned* lk = (unsigned*)Kt_s[cur];
      for (int i = threadIdx.x; i < 32 * DK / 2; i += 128) lk[i] = gk[i];
      unsigned* lv = (unsigned*)Vs[cur];
      for (int i = threadIdx.x; i < DK * 32 / 2; i += 128) {
        int d = i / 16, kk = (i % 16) * 2;
        const __bf16* gv = Vp + (size_t)d * SLEN + key0 + kk;
        lv[i] = *(const unsigned*)gv;
      }
    }
    __syncthreads();
    compute_tile(cur, key0);
  }
#endif

  const int b = bh >> 4, h = bh & 15;
#pragma unroll
  for (int nt = 0; nt < 4; ++nt) {
    int d = nt * 16 + nl;
#pragma unroll
    for (int r = 0; r < 8; ++r) {
      int q = q0 + r + hi * 8;
      float v = o[nt][r] / lrow[r];
      AO[((size_t)b * SLEN + q) * DMODEL + h * DK + d] = (__bf16)v;
    }
  }
}

// ---- launch ----------------------------------------------------------------
extern "C" void kernel_launch(void* const* d_in, const int* in_sizes, int n_in,
                              void* d_out, int out_size, void* d_ws, size_t ws_size,
                              hipStream_t stream) {
  (void)in_sizes; (void)n_in; (void)out_size; (void)ws_size;
  const float* x   = (const float*)d_in[0];
  const int*   pos = (const int*)  d_in[1];
  const float* wq  = (const float*)d_in[2];
  const float* wk  = (const float*)d_in[3];
  const float* wv  = (const float*)d_in[4];
  const float* wo  = (const float*)d_in[5];

  char* ws = (char*)d_ws;
  const size_t XB = (size_t)NB * SLEN * DMODEL * 2;  // 8 MB
  const size_t WB = (size_t)DMODEL * DMODEL * 2;     // 2 MB
  __bf16* xb  = (__bf16*)(ws);
  __bf16* wqb = (__bf16*)(ws + XB);
  __bf16* wkb = (__bf16*)(ws + XB + WB);
  __bf16* wvb = (__bf16*)(ws + XB + 2 * WB);
  __bf16* wob = (__bf16*)(ws + XB + 3 * WB);
  __bf16* Qw  = (__bf16*)(ws + XB + 4 * WB);
  __bf16* Kw  = (__bf16*)(ws + 2 * XB + 4 * WB);
  __bf16* Vw  = (__bf16*)(ws + 3 * XB + 4 * WB);
  __bf16* AO  = (__bf16*)(ws + 4 * XB + 4 * WB);

  const int nX = NB * SLEN * DMODEL;
  const int nW = DMODEL * DMODEL;
  cvt_f32_bf16_x4<<<nX / 1024, 256, 0, stream>>>((const float4*)x,  (v4bf*)xb,  nX / 4);
  cvt_f32_bf16_x4<<<nW / 1024, 256, 0, stream>>>((const float4*)wq, (v4bf*)wqb, nW / 4);
  cvt_f32_bf16_x4<<<nW / 1024, 256, 0, stream>>>((const float4*)wk, (v4bf*)wkb, nW / 4);
  cvt_f32_bf16_x4<<<nW / 1024, 256, 0, stream>>>((const float4*)wv, (v4bf*)wvb, nW / 4);
  cvt_f32_bf16_x4<<<nW / 1024, 256, 0, stream>>>((const float4*)wo, (v4bf*)wob, nW / 4);

  gemm_wmma<<<512, 256, 0, stream>>>(xb, wqb, (void*)Qw, pos, 0);
  gemm_wmma<<<512, 256, 0, stream>>>(xb, wkb, (void*)Kw, pos, 1);
  gemm_wmma<<<512, 256, 0, stream>>>(xb, wvb, (void*)Vw, pos, 2);

  attn_flash<<<NB * NH * (SLEN / 64), 128, 0, stream>>>(Qw, Kw, Vw, AO);

  gemm_wmma<<<512, 256, 0, stream>>>(AO, wob, d_out, pos, 3);
}
